// InteractionDecoder_68504728371698
// MI455X (gfx1250) — compile-verified, matrix operand order checked by reading, not run on hardware
//
#include <hip/hip_runtime.h>
#include <stdint.h>

#define LATENT 128
#define TILE_M 128   // edges per block

typedef __attribute__((ext_vector_type(8)))  float          v8f;
typedef __attribute__((ext_vector_type(4)))  float          v4f;
typedef __attribute__((ext_vector_type(16))) __bf16         v16bf;
typedef __attribute__((ext_vector_type(4)))  unsigned short v4us;
typedef __attribute__((ext_vector_type(8)))  unsigned short v8us;
typedef __attribute__((ext_vector_type(16))) unsigned short v16us;

__device__ __forceinline__ unsigned short f2bf(float f) {
  unsigned u = __builtin_bit_cast(unsigned, f);
  u += 0x7FFFu + ((u >> 16) & 1u);   // round-to-nearest-even
  return (unsigned short)(u >> 16);
}
__device__ __forceinline__ float bf2f(unsigned short h) {
  unsigned u = ((unsigned)h) << 16;
  return __builtin_bit_cast(float, u);
}

// ---------------------------------------------------------------------------
__global__ void zero_kernel(float* p, int n) {
  int i = blockIdx.x * blockDim.x + threadIdx.x;
  if (i < n) p[i] = 0.0f;
}

// Pack six 128x128 fp32 weights into bf16 WMMA B-fragment order.
// packed[((kt*8+nt)*32 + lane)*16 + j] = bf16(W[k][n]),
//   kbase = kt*32 + (lane>=16 ? 8 : 0); k = kbase + j (j<8) / kbase+16+(j-8)
//   n = nt*16 + (lane&15)
__global__ void pack_weights_kernel(const float* w0, const float* w1,
                                    const float* w2, const float* w3,
                                    const float* w4, const float* w5,
                                    unsigned short* packed) {
  int idx = blockIdx.x * blockDim.x + threadIdx.x;
  if (idx >= 6 * 16384) return;
  int widx = idx >> 14;
  int o    = idx & 16383;
  int j    = o & 15;
  int lane = (o >> 4) & 31;
  int nt   = (o >> 9) & 7;
  int kt   = o >> 12;
  int kbase = kt * 32 + ((lane & 16) ? 8 : 0);
  int k     = kbase + j + ((j >= 8) ? 8 : 0);
  int n     = nt * 16 + (lane & 15);
  const float* W = (widx == 0) ? w0 : (widx == 1) ? w1 : (widx == 2) ? w2
                 : (widx == 3) ? w3 : (widx == 4) ? w4 : w5;
  packed[idx] = f2bf(W[k * LATENT + n]);
}

// ---------------------------------------------------------------------------
struct MlpArgs {
  const float* X;                 // interaction_latent [E,128]
  const unsigned short* packed;   // 6 * 16384 bf16 weights (B-fragment order)
  const float* biasL[6];          // per-layer biases (b1,b2 per MLP)
  const float* W3[3];             // [128, dout] row-major heads
  const float* b3[3];
  const int*   eidx;              // [2,E]
  const float* edge_attr;         // [E,3]
  const int*   node_type;         // [N,4]
  const float* va; const float* vb; const float* vc;   // [E,3]
  float* out_f;                   // [E,3]
  float* out_a;                   // [E,3]
  float* lam;                     // [E]
  float* sum_f; float* sum_a; float* cnt;              // node accumulators
  int E;
};

__launch_bounds__(256)
__global__ void mlp_edge_kernel(MlpArgs A) {
  __shared__ unsigned short Xs[TILE_M * LATENT];   // 32 KB bf16 latent tile
  __shared__ unsigned short Hs[TILE_M * LATENT];   // 32 KB hidden (in-place reuse)
  __shared__ unsigned short Wb[2][16384];          // 64 KB double-buffered weights

  const int tid  = threadIdx.x;
  const int lane = tid & 31;
  const int wv   = tid >> 5;
  const int e0   = blockIdx.x * TILE_M;

  // ---- stage latent tile -> LDS (bf16); float4 loads, b64 stores ----
  for (int i4 = tid; i4 < TILE_M * LATENT / 4; i4 += 256) {
    int row = i4 >> 5;                       // 32 float4 per row
    int e = e0 + row;
    v4f x = {0.f, 0.f, 0.f, 0.f};
    if (e < A.E) x = *(const v4f*)&A.X[(size_t)e * LATENT + (i4 & 31) * 4];
    v4us h = { f2bf(x.x), f2bf(x.y), f2bf(x.z), f2bf(x.w) };
    *(v4us*)&Xs[i4 * 4] = h;
  }

  // cooperative 32KB weight copy: global (L2-hot) -> LDS
  auto copy_w = [&](const unsigned short* gsrc, unsigned short* dst) {
#pragma unroll
    for (int v = 0; v < 8; ++v) {
      int i = tid + v * 256;
      ((v8us*)dst)[i] = ((const v8us*)gsrc)[i];
    }
  };

  copy_w(A.packed, Wb[0]);
  __syncthreads();

  const int aRow  = wv * 16 + (lane & 15);          // A-matrix row (M) for this lane
  const int kHalf = (lane & 16) ? 8 : 0;            // K split per half-wave
  const int mrow0 = wv * 16 + ((lane & 16) ? 8 : 0);// C/D row for VGPR r=0
  const int ncol0 = lane & 15;

  // one 16(M) x 128(N) x 128(K) GEMM slice per wave; wave-private M-slice
  auto run_gemm = [&](const unsigned short* S, unsigned short* D,
                      const unsigned short* Wl, const float* bias) {
    v8f acc[8] = {};
#pragma unroll
    for (int kt = 0; kt < 4; ++kt) {
      int kb = kt * 32 + kHalf;
      v8us alo = *(const v8us*)&S[aRow * LATENT + kb];        // ds_load_b128
      v8us ahi = *(const v8us*)&S[aRow * LATENT + kb + 16];
      v16bf a  = __builtin_bit_cast(v16bf, __builtin_shufflevector(
          alo, ahi, 0,1,2,3,4,5,6,7,8,9,10,11,12,13,14,15));
#pragma unroll
      for (int nt = 0; nt < 8; ++nt) {
        const unsigned short* bp = Wl + (((kt * 8 + nt) * 32 + lane) << 4);
        v8us blo = *(const v8us*)bp;                          // ds_load_b128
        v8us bhi = *(const v8us*)(bp + 8);
        v16bf b  = __builtin_bit_cast(v16bf, __builtin_shufflevector(
            blo, bhi, 0,1,2,3,4,5,6,7,8,9,10,11,12,13,14,15));
        acc[nt] = __builtin_amdgcn_wmma_f32_16x16x32_bf16(
            false, a, false, b, (short)0, acc[nt], false, false);
      }
    }
    // bias + relu + store bf16 into D (wave-private rows)
#pragma unroll
    for (int nt = 0; nt < 8; ++nt) {
      int n = nt * 16 + ncol0;
      float bn = bias[n];
#pragma unroll
      for (int r = 0; r < 8; ++r) {
        float v = fmaxf(acc[nt][r] + bn, 0.0f);
        D[(mrow0 + r) * LATENT + n] = f2bf(v);
      }
    }
  };

  float cf0 = 0, cf1 = 0, cf2 = 0, ca0 = 0, ca1 = 0, ca2 = 0, lm = 0;

  for (int l = 0; l < 6; ++l) {
    if (l < 5) copy_w(A.packed + (size_t)(l + 1) * 16384, Wb[(l + 1) & 1]);
    const unsigned short* src = (l & 1) ? Hs : Xs;
    run_gemm(src, Hs, Wb[l & 1], A.biasL[l]);
    __syncthreads();                 // Hs fully written; next weight buffer staged
    if (l & 1) {
      int m = l >> 1;
      if (tid < TILE_M) {            // tiny head: 128x{3,3,1}, VALU dots
        const float* W3 = A.W3[m];
        const float* b3 = A.b3[m];
        int dout = (m == 2) ? 1 : 3;
        float o0 = b3[0];
        float o1 = (dout > 1) ? b3[1] : 0.0f;
        float o2 = (dout > 1) ? b3[2] : 0.0f;
        for (int k = 0; k < LATENT; ++k) {
          float h = bf2f(Hs[tid * LATENT + k]);
          o0 += h * W3[k * dout + 0];
          if (dout > 1) {
            o1 += h * W3[k * dout + 1];
            o2 += h * W3[k * dout + 2];
          }
        }
        if (m == 0)      { cf0 = o0; cf1 = o1; cf2 = o2; }
        else if (m == 1) { ca0 = o0; ca1 = o1; ca2 = o2; }
        else             { lm = o0; }
      }
      __syncthreads();               // head reads done before Hs is overwritten
    }
  }

  // ---- per-edge assembly + masked scatter-add for group means ----
  int e = e0 + tid;
  if (tid < TILE_M && e < A.E) {
    float f[3], aa[3];
#pragma unroll
    for (int i = 0; i < 3; ++i) {
      float x = A.va[(size_t)e * 3 + i];
      float y = A.vb[(size_t)e * 3 + i];
      float z = A.vc[(size_t)e * 3 + i];
      f[i]  = cf0 * x + cf1 * y + cf2 * z;
      aa[i] = ca0 * x + ca1 * y + ca2 * z;
    }
    int s = A.eidx[e];
    int r = A.eidx[(size_t)A.E + e];
    bool virt = (A.edge_attr[(size_t)e * 3] == -1.0f);
    bool rg   = (A.node_type[(size_t)r * 4 + 3] == -1);
    bool sg   = (A.node_type[(size_t)s * 4 + 3] == -1);
    bool mask = virt && rg && !sg;
#pragma unroll
    for (int i = 0; i < 3; ++i) {
      A.out_f[(size_t)e * 3 + i] = f[i];
      A.out_a[(size_t)e * 3 + i] = aa[i];
    }
    A.lam[e] = lm;
    if (mask) {
#pragma unroll
      for (int i = 0; i < 3; ++i) {
        atomicAdd(&A.sum_f[(size_t)r * 3 + i], f[i]);
        atomicAdd(&A.sum_a[(size_t)r * 3 + i], aa[i]);
      }
      atomicAdd(&A.cnt[r], 1.0f);
    }
  }
}

// ---------------------------------------------------------------------------
struct FinArgs {
  const int* eidx; const float* edge_attr; const int* node_type;
  const float* spos; const float* rpos;
  const float* sum_f; const float* sum_a; const float* cnt;
  const float* lam;
  float* out_f; float* out_a;
  int E;
};

__global__ void finalize_kernel(FinArgs A) {
  int e = blockIdx.x * blockDim.x + threadIdx.x;
  if (e >= A.E) return;
  int s = A.eidx[e];
  int r = A.eidx[(size_t)A.E + e];
  bool virt = (A.edge_attr[(size_t)e * 3] == -1.0f);
  bool rg   = (A.node_type[(size_t)r * 4 + 3] == -1);
  bool sg   = (A.node_type[(size_t)s * 4 + 3] == -1);
  bool mask = virt && rg && !sg;

  float f[3], aa[3];
#pragma unroll
  for (int i = 0; i < 3; ++i) {
    f[i]  = A.out_f[(size_t)e * 3 + i];
    aa[i] = A.out_a[(size_t)e * 3 + i];
  }
  if (mask) {
    float c = fmaxf(A.cnt[r], 1.0f);
#pragma unroll
    for (int i = 0; i < 3; ++i) {
      f[i]  -= A.sum_f[(size_t)r * 3 + i] / c;
      aa[i] -= A.sum_a[(size_t)r * 3 + i] / c;
    }
  }
  float lm = A.lam[e];
  float lv[3], g[3];
#pragma unroll
  for (int i = 0; i < 3; ++i) {
    lv[i] = A.spos[(size_t)e * 3 + i] - A.rpos[(size_t)e * 3 + i];
    g[i]  = f[i] * lm;
  }
  float m0 = lv[1] * g[2] - lv[2] * g[1];
  float m1 = lv[2] * g[0] - lv[0] * g[2];
  float m2 = lv[0] * g[1] - lv[1] * g[0];
  A.out_f[(size_t)e * 3 + 0] = f[0];
  A.out_f[(size_t)e * 3 + 1] = f[1];
  A.out_f[(size_t)e * 3 + 2] = f[2];
  A.out_a[(size_t)e * 3 + 0] = aa[0] - m0;
  A.out_a[(size_t)e * 3 + 1] = aa[1] - m1;
  A.out_a[(size_t)e * 3 + 2] = aa[2] - m2;
}

// ---------------------------------------------------------------------------
extern "C" void kernel_launch(void* const* d_in, const int* in_sizes, int n_in,
                              void* d_out, int out_size, void* d_ws, size_t ws_size,
                              hipStream_t stream) {
  const int*   edge_index = (const int*)d_in[0];
  const float* edge_attr  = (const float*)d_in[1];
  const float* spos       = (const float*)d_in[2];
  const float* rpos       = (const float*)d_in[3];
  const float* va         = (const float*)d_in[4];
  const float* vb         = (const float*)d_in[5];
  const float* vc         = (const float*)d_in[6];
  const float* lat        = (const float*)d_in[7];
  /* d_in[8] = w_nodes (unused by reference output) */
  const int*   node_type  = (const int*)d_in[9];
  // params pytree flattened with sorted dict keys: "fs", "i1", "i2";
  // each is [(w1,b1),(w2,b2),(w3,b3)] -> 6 leaves.
  const float* fsW1 = (const float*)d_in[10]; const float* fsB1 = (const float*)d_in[11];
  const float* fsW2 = (const float*)d_in[12]; const float* fsB2 = (const float*)d_in[13];
  const float* fsW3 = (const float*)d_in[14]; const float* fsB3 = (const float*)d_in[15];
  const float* i1W1 = (const float*)d_in[16]; const float* i1B1 = (const float*)d_in[17];
  const float* i1W2 = (const float*)d_in[18]; const float* i1B2 = (const float*)d_in[19];
  const float* i1W3 = (const float*)d_in[20]; const float* i1B3 = (const float*)d_in[21];
  const float* i2W1 = (const float*)d_in[22]; const float* i2B1 = (const float*)d_in[23];
  const float* i2W2 = (const float*)d_in[24]; const float* i2B2 = (const float*)d_in[25];
  const float* i2W3 = (const float*)d_in[26]; const float* i2B3 = (const float*)d_in[27];

  const int E = in_sizes[4] / 3;    // vector_a is [E,3]
  const int N = in_sizes[9] / 4;    // node_type is [N,4]

  // workspace layout (floats): sum_f[3N] | sum_a[3N] | cnt[N] | lam[E] | packedW(bf16)
  float* ws    = (float*)d_ws;
  float* sum_f = ws;
  float* sum_a = ws + (size_t)3 * N;
  float* cnt   = ws + (size_t)6 * N;
  float* lam   = ws + (size_t)7 * N;
  unsigned short* packed = (unsigned short*)(ws + (size_t)7 * N + E);

  int zn = 7 * N;
  zero_kernel<<<(zn + 255) / 256, 256, 0, stream>>>(ws, zn);
  pack_weights_kernel<<<(6 * 16384 + 255) / 256, 256, 0, stream>>>(
      i1W1, i1W2, i2W1, i2W2, fsW1, fsW2, packed);

  MlpArgs MA;
  MA.X = lat; MA.packed = packed;
  MA.biasL[0] = i1B1; MA.biasL[1] = i1B2;
  MA.biasL[2] = i2B1; MA.biasL[3] = i2B2;
  MA.biasL[4] = fsB1; MA.biasL[5] = fsB2;
  MA.W3[0] = i1W3; MA.W3[1] = i2W3; MA.W3[2] = fsW3;
  MA.b3[0] = i1B3; MA.b3[1] = i2B3; MA.b3[2] = fsB3;
  MA.eidx = edge_index; MA.edge_attr = edge_attr; MA.node_type = node_type;
  MA.va = va; MA.vb = vb; MA.vc = vc;
  MA.out_f = (float*)d_out;
  MA.out_a = (float*)d_out + (size_t)3 * E;
  MA.lam = lam; MA.sum_f = sum_f; MA.sum_a = sum_a; MA.cnt = cnt; MA.E = E;
  int nblk = (E + TILE_M - 1) / TILE_M;
  mlp_edge_kernel<<<nblk, 256, 0, stream>>>(MA);

  FinArgs FA;
  FA.eidx = edge_index; FA.edge_attr = edge_attr; FA.node_type = node_type;
  FA.spos = spos; FA.rpos = rpos;
  FA.sum_f = sum_f; FA.sum_a = sum_a; FA.cnt = cnt; FA.lam = lam;
  FA.out_f = (float*)d_out;
  FA.out_a = (float*)d_out + (size_t)3 * E;
  FA.E = E;
  finalize_kernel<<<(E + 255) / 256, 256, 0, stream>>>(FA);
}
